// GCN_17910013624553
// MI455X (gfx1250) — matrix operand order repaired and checked by reading, not compiled, not run on hardware
//
#include <hip/hip_runtime.h>
#include <hip/hip_bf16.h>

// ---------------------------------------------------------------------------
// GCN inference pipeline for MI455X (gfx1250), wave32 + WMMA bf16.
//   h = relu(BN(spmm(A, X@W0 + b0) + bias0))
//   h = relu(BN(spmm(A, h@W1) + bias1))
//   out = (h@Wout + bout)[idx]
// All GEMMs: v_wmma_f32_16x16x32_bf16 (bf16 in, fp32 accumulate).
// Activations kept in bf16 so the 100000x512 gather source fits in the
// 192 MB L2 (102.4 MB), halving SPMM gather traffic vs fp32.
// Wave tile 32x64 (2x4 WMMA fragments): 12 fragment loads per 8 WMMAs.
// All per-K-step fragments (a0,a1,b[0..3]) are loaded into distinct
// registers before the WMMA block so the compiler can drain them with
// partial s_wait_loadcnt values instead of full waits.
// ---------------------------------------------------------------------------

#define GCN_N     100000
#define GCN_E     3200000
#define GCN_DIN   128
#define GCN_H     512
#define GCN_EMB   64
#define GCN_M     50000

typedef __attribute__((ext_vector_type(16))) __bf16 bf16x16;
typedef __attribute__((ext_vector_type(8)))  float  f32x8;

union FragU { bf16x16 v; uint4 q[2]; };

// ---- WMMA fragment loaders -------------------------------------------------
// A: row-major [M,K] bf16. 16-bit A 16x32 layout:
//   lane L: m = L&15; lanes 0-15 hold K={0..7, 16..23}, lanes 16-31 K={8..15, 24..31}
__device__ __forceinline__ bf16x16 load_a_frag(const __hip_bfloat16* __restrict__ A,
                                               int lda, int m0, int k0, int lane) {
    const int m  = lane & 15;
    const int kb = (lane >> 4) << 3;            // 0 or 8
    const __hip_bfloat16* p = A + (size_t)(m0 + m) * lda + (k0 + kb);
    FragU f;
    f.q[0] = *(const uint4*)(p);                // K = k0+kb   .. +7   (16B)
    f.q[1] = *(const uint4*)(p + 16);           // K = k0+16+kb.. +7   (16B)
    return f.v;
}

// B: stored transposed, K-contiguous: BT[n][k], row stride = K.
// 16-bit B 32x16 layout: lane L: n = L&15; lanes 0-15 hold K={0..7, 16..23}, lanes 16-31 K={8..15, 24..31}
__device__ __forceinline__ bf16x16 load_b_frag(const __hip_bfloat16* __restrict__ BT,
                                               int ldb, int n0, int k0, int lane) {
    const int n  = lane & 15;
    const int kb = (lane >> 4) << 3;            // 0 or 8
    const __hip_bfloat16* p = BT + (size_t)(n0 + n) * ldb + (k0 + kb);
    FragU f;
    f.q[0] = *(const uint4*)(p);                // 16B: K = kb..kb+7
    f.q[1] = *(const uint4*)(p + 16);           // 16B: K = 16+kb..16+kb+7
    return f.v;
}

// ---- WMMA GEMM: one wave per 32x64 output tile, 2x4 fragment grid ---------
// C = A[M,K](bf16) * BT[N,K]^T (bf16) + bias, fp32 accumulate.
// grid = (Ncols/64, Mrows/32), block = 32 (single wave; EXEC all-ones).
template <bool STORE_BF16>
__global__ void wmma_gemm_bf16(const __hip_bfloat16* __restrict__ A,
                               const __hip_bfloat16* __restrict__ BT,
                               const float* __restrict__ bias,
                               void* __restrict__ Cout,
                               int K, int ldc) {
    const int lane = threadIdx.x & 31;
    const int m0 = blockIdx.y * 32;
    const int n0 = blockIdx.x * 64;

    f32x8 acc[2][4] = {};

    for (int k0 = 0; k0 < K; k0 += 32) {
        // Issue all 12 fragment loads up front (distinct registers) so the
        // WMMA block below can consume them with partial loadcnt waits.
        const bf16x16 a0 = load_a_frag(A, K, m0,      k0, lane);
        const bf16x16 a1 = load_a_frag(A, K, m0 + 16, k0, lane);
        bf16x16 bfr[4];
        #pragma unroll
        for (int nt = 0; nt < 4; ++nt)
            bfr[nt] = load_b_frag(BT, K, n0 + nt * 16, k0, lane);

        #pragma unroll
        for (int nt = 0; nt < 4; ++nt) {
            acc[0][nt] = __builtin_amdgcn_wmma_f32_16x16x32_bf16(
                false, a0, false, bfr[nt], (short)0, acc[0][nt], false, false);
            acc[1][nt] = __builtin_amdgcn_wmma_f32_16x16x32_bf16(
                false, a1, false, bfr[nt], (short)0, acc[1][nt], false, false);
        }
    }

    // C 16x16 f32 layout: VGPR r, lanes 0-15 -> M=r, lanes 16-31 -> M=r+8; N = lane&15
    const int hi = (lane >> 4) * 8;
    const int nl = lane & 15;
    #pragma unroll
    for (int mt = 0; mt < 2; ++mt) {
        #pragma unroll
        for (int nt = 0; nt < 4; ++nt) {
            const int col = n0 + nt * 16 + nl;
            const float bv = bias ? bias[col] : 0.0f;
            #pragma unroll
            for (int r = 0; r < 8; ++r) {
                const int row = m0 + mt * 16 + hi + r;
                const float v = acc[mt][nt][r] + bv;
                if (STORE_BF16)
                    ((__hip_bfloat16*)Cout)[(size_t)row * ldc + col] = __float2bfloat16(v);
                else
                    ((float*)Cout)[(size_t)row * ldc + col] = v;
            }
        }
    }
}

// ---- SPMM: COO segment-sum via fp32 atomics, bf16 gather source -----------
// out[r,:] += val[e] * Hin[col[e],:] ; 64 threads per edge, 8 features each.
__global__ void spmm_coo_atomic(const int* __restrict__ er, const int* __restrict__ ec,
                                const float* __restrict__ ev,
                                const __hip_bfloat16* __restrict__ Hin,
                                float* __restrict__ Hacc, int E) {
    const long long gid = (long long)blockIdx.x * blockDim.x + threadIdx.x;
    const int e = (int)(gid >> 6);
    if (e >= E) return;
    const int f0 = ((int)(gid & 63)) * 8;
    const int r = er[e];
    const int c = ec[e];
    const float v = ev[e];
    const uint4 q = *(const uint4*)(Hin + (size_t)c * GCN_H + f0);  // 8 bf16, 16B
    const __hip_bfloat16* hb = (const __hip_bfloat16*)&q;
    float* dst = Hacc + (size_t)r * GCN_H + f0;
    #pragma unroll
    for (int i = 0; i < 8; ++i)
        atomicAdd(dst + i, v * __bfloat162float(hb[i]));            // global_atomic_add_f32
}

// ---- fused (+bias) -> BN(inference) -> ReLU -> bf16 -----------------------
__global__ void bias_bn_relu_to_bf16(const float* __restrict__ X,
                                     const float* __restrict__ biasAdd,
                                     const float* __restrict__ g,
                                     const float* __restrict__ be,
                                     const float* __restrict__ mu,
                                     const float* __restrict__ var,
                                     __hip_bfloat16* __restrict__ Y,
                                     long long total) {
    const long long i = (long long)blockIdx.x * blockDim.x + threadIdx.x;
    if (i >= total) return;
    const int h = (int)(i & (GCN_H - 1));
    const float x = X[i] + biasAdd[h];
    float y = g[h] * (x - mu[h]) * rsqrtf(var[h] + 1e-5f) + be[h];
    y = y > 0.0f ? y : 0.0f;
    Y[i] = __float2bfloat16(y);
}

// ---- small utility kernels -------------------------------------------------
__global__ void zero_f32(float* __restrict__ p, long long n) {
    const long long i = (long long)blockIdx.x * blockDim.x + threadIdx.x;
    if (i < n) p[i] = 0.0f;
}

__global__ void cvt_f32_to_bf16(const float* __restrict__ s, __hip_bfloat16* __restrict__ d,
                                long long n) {
    const long long i = (long long)blockIdx.x * blockDim.x + threadIdx.x;
    if (i < n) d[i] = __float2bfloat16(s[i]);
}

// W[rows][cols] (fp32, row-major) -> WT[cols][rows] (bf16, K-contiguous)
__global__ void transpose_to_bf16(const float* __restrict__ W, __hip_bfloat16* __restrict__ WT,
                                  int rows, int cols) {
    const long long i = (long long)blockIdx.x * blockDim.x + threadIdx.x;
    if (i >= (long long)rows * cols) return;
    const int rr = (int)(i / cols);
    const int cc = (int)(i % cols);
    WT[(size_t)cc * rows + rr] = __float2bfloat16(W[i]);
}

// out[m,:] = src[idx[m],:]  (EMB = 64 floats per row)
__global__ void gather_rows(const float* __restrict__ src, const int* __restrict__ idx,
                            float* __restrict__ dst, int M) {
    const long long gid = (long long)blockIdx.x * blockDim.x + threadIdx.x;
    const int m = (int)(gid >> 6);
    if (m >= M) return;
    const int j = (int)(gid & 63);
    dst[(size_t)m * GCN_EMB + j] = src[(size_t)idx[m] * GCN_EMB + j];
}

// ---------------------------------------------------------------------------
extern "C" void kernel_launch(void* const* d_in, const int* in_sizes, int n_in,
                              void* d_out, int out_size, void* d_ws, size_t ws_size,
                              hipStream_t stream) {
    (void)in_sizes; (void)n_in; (void)out_size; (void)ws_size;

    const float* features = (const float*)d_in[0];
    const int*   edge_row = (const int*)  d_in[1];
    const int*   edge_col = (const int*)  d_in[2];
    const float* edge_val = (const float*)d_in[3];
    const int*   idx      = (const int*)  d_in[4];
    const float* W0    = (const float*)d_in[5];
    const float* b0    = (const float*)d_in[6];
    const float* bias0 = (const float*)d_in[7];
    const float* g0    = (const float*)d_in[8];
    const float* be0   = (const float*)d_in[9];
    const float* m0    = (const float*)d_in[10];
    const float* v0    = (const float*)d_in[11];
    const float* W1    = (const float*)d_in[12];
    const float* bias1 = (const float*)d_in[13];
    const float* g1    = (const float*)d_in[14];
    const float* be1   = (const float*)d_in[15];
    const float* m1    = (const float*)d_in[16];
    const float* v1    = (const float*)d_in[17];
    const float* Wout  = (const float*)d_in[18];
    const float* bout  = (const float*)d_in[19];
    float* out = (float*)d_out;

    // -------- workspace carve-up (256B aligned) --------
    char*  base = (char*)d_ws;
    size_t off  = 0;
    auto carve = [&](size_t bytes) -> void* {
        void* p = base + off;
        off = (off + bytes + 255) & ~(size_t)255;
        return p;
    };
    __hip_bfloat16* featbf = (__hip_bfloat16*)carve((size_t)GCN_N * GCN_DIN * 2);
    __hip_bfloat16* w0t    = (__hip_bfloat16*)carve((size_t)GCN_H * GCN_DIN * 2);   // [H][DIN]
    __hip_bfloat16* w1t    = (__hip_bfloat16*)carve((size_t)GCN_H * GCN_H   * 2);   // [H][H]
    __hip_bfloat16* woutt  = (__hip_bfloat16*)carve((size_t)GCN_EMB * GCN_H * 2);   // [EMB][H]
    __hip_bfloat16* hA     = (__hip_bfloat16*)carve((size_t)GCN_N * GCN_H * 2);     // gemm out / bn out
    __hip_bfloat16* hC     = (__hip_bfloat16*)carve((size_t)GCN_N * GCN_H * 2);     // layer-1 gemm out
    float*          hB     = (float*)         carve((size_t)GCN_N * GCN_H * 4);     // spmm fp32 accum
    float*          outf   = (float*)         carve((size_t)GCN_N * GCN_EMB * 4);   // pre-gather

    const long long NH = (long long)GCN_N * GCN_H;

    // -------- prep: bf16 convert + weight transposes --------
    {
        long long n = (long long)GCN_N * GCN_DIN;
        cvt_f32_to_bf16<<<(unsigned)((n + 255) / 256), 256, 0, stream>>>(features, featbf, n);
    }
    transpose_to_bf16<<<(GCN_DIN * GCN_H + 255) / 256, 256, 0, stream>>>(W0, w0t, GCN_DIN, GCN_H);
    transpose_to_bf16<<<(GCN_H * GCN_H + 255) / 256, 256, 0, stream>>>(W1, w1t, GCN_H, GCN_H);
    transpose_to_bf16<<<(GCN_H * GCN_EMB + 255) / 256, 256, 0, stream>>>(Wout, woutt, GCN_H, GCN_EMB);

    const dim3 gemmGridH(GCN_H / 64, GCN_N / 32);     // (8, 3125)
    const dim3 gemmGridE(GCN_EMB / 64, GCN_N / 32);   // (1, 3125)
    const unsigned spmmBlocks = (unsigned)(((long long)GCN_E * 64 + 255) / 256);
    const unsigned ewBlocks   = (unsigned)((NH + 255) / 256);

    // -------- layer 0 --------
    wmma_gemm_bf16<true><<<gemmGridH, 32, 0, stream>>>(featbf, w0t, b0, hA, GCN_DIN, GCN_H);
    zero_f32<<<ewBlocks, 256, 0, stream>>>(hB, NH);
    spmm_coo_atomic<<<spmmBlocks, 256, 0, stream>>>(edge_row, edge_col, edge_val, hA, hB, GCN_E);
    bias_bn_relu_to_bf16<<<ewBlocks, 256, 0, stream>>>(hB, bias0, g0, be0, m0, v0, hA, NH);

    // -------- layer 1 --------
    wmma_gemm_bf16<true><<<gemmGridH, 32, 0, stream>>>(hA, w1t, nullptr, hC, GCN_H, GCN_H);
    zero_f32<<<ewBlocks, 256, 0, stream>>>(hB, NH);
    spmm_coo_atomic<<<spmmBlocks, 256, 0, stream>>>(edge_row, edge_col, edge_val, hC, hB, GCN_E);
    bias_bn_relu_to_bf16<<<ewBlocks, 256, 0, stream>>>(hB, bias1, g1, be1, m1, v1, hA, NH);

    // -------- output projection + gather --------
    wmma_gemm_bf16<false><<<gemmGridE, 32, 0, stream>>>(hA, woutt, bout, outf, GCN_H, GCN_EMB);
    gather_rows<<<(unsigned)(((long long)GCN_M * GCN_EMB + 255) / 256), 256, 0, stream>>>(
        outf, idx, out, GCN_M);
}
